// GNN_GINEConv_7275674600532
// MI455X (gfx1250) — matrix operand order repaired and checked by reading, not compiled, not run on hardware
//
#include <hip/hip_runtime.h>
#include <hip/hip_bf16.h>
#include <math.h>

// GINE GNN inference for MI455X (gfx1250, wave32).
// Workspace requirement: 2*N*H*4 + 2*L*H*H*2 + G*(H+1)*4 + G*(H/2)*4 bytes (~103 MB).

#define H 128

typedef __attribute__((ext_vector_type(16))) __bf16 v16bf;
typedef __attribute__((ext_vector_type(8)))  float  v8f;
typedef int v4i_vs __attribute__((vector_size(16)));   // pointee type of async-LDS builtin

union ABu { v16bf v; unsigned int u[8]; };

#if defined(__has_builtin)
#if __has_builtin(__builtin_amdgcn_global_load_async_to_lds_b128) && \
    __has_builtin(__builtin_amdgcn_s_wait_asynccnt)
#define USE_ASYNC_LDS 1
#endif
#endif

// XOR swizzle at 16-byte (8-element) granularity: removes the 256B-period bank
// conflicts of row-strided fragment loads, keeps >=8-element vectors contiguous.
__device__ __forceinline__ int swz(int row, int k) { return k ^ ((row & 15) << 3); }

// Stage a 128x128 bf16 W^T tile into LDS. Prefer the CDNA5 async DMA path
// (global_load_async_to_lds_b128, ASYNCcnt) over the VGPR round trip.
__device__ __forceinline__ void stage_weights(const __bf16* __restrict__ wsrc,
                                              __bf16* sW, int tid) {
#ifdef USE_ASYNC_LDS
  for (int i = tid; i < (H * H) / 8; i += 128) {
    int n = (i * 8) >> 7, k = (i * 8) & (H - 1);
    __builtin_amdgcn_global_load_async_to_lds_b128(
        (__attribute__((address_space(1))) v4i_vs*)(void*)(const_cast<__bf16*>(wsrc) + n * H + k),
        (__attribute__((address_space(3))) v4i_vs*)(void*)&sW[n * H + swz(n, k)],
        0, 0);
  }
  __builtin_amdgcn_s_wait_asynccnt(0);
#else
  for (int i = tid; i < (H * H) / 8; i += 128) {
    int n = (i * 8) >> 7, k = (i * 8) & (H - 1);
    *(uint4*)&sW[n * H + swz(n, k)] = *(const uint4*)(wsrc + n * H + k);
  }
#endif
}

// ---------------------------------------------------------------------------
// h[n,c] = x[n]*enc_w[c] + enc_b[c]      (rank-1 encoder, x is [N,1])
// ---------------------------------------------------------------------------
__global__ void encode_kernel(const float* __restrict__ x,
                              const float* __restrict__ ew,
                              const float* __restrict__ eb,
                              float* __restrict__ h, int nN) {
  long long t = (long long)blockIdx.x * blockDim.x + threadIdx.x;
  if (t >= (long long)nN * H) return;
  int n = (int)(t >> 7), c = (int)(t & (H - 1));
  h[t] = x[n] * ew[c] + eb[c];
}

// ---------------------------------------------------------------------------
// Pre-convert GIN MLP weights to bf16, transposed to [n][k] so B-fragment
// K-pairs are contiguous. Output layout: mat = 2*l + (0:w1, 1:w2).
// ---------------------------------------------------------------------------
__global__ void prep_w_kernel(const float* __restrict__ w1,
                              const float* __restrict__ w2,
                              __bf16* __restrict__ wt, int L) {
  int i = blockIdx.x * blockDim.x + threadIdx.x;
  if (i >= 2 * L * H * H) return;
  int mat = i >> 14;                 // H*H == 16384
  int r   = i & (H * H - 1);
  int n   = r >> 7, k = r & (H - 1);
  int l   = mat >> 1;
  const float* src = (mat & 1) ? w2 : w1;
  wt[i] = (__bf16)src[l * H * H + k * H + n];
}

// ---------------------------------------------------------------------------
// One wave per edge; lane owns 4 channels. msg = relu(h[src] + ea), ea
// recomputed from the scalar edge_attr (saves 819 MB of [E,H] traffic).
// Scatter-add with global f32 atomics.
// ---------------------------------------------------------------------------
__global__ __launch_bounds__(256) void edge_msg_kernel(
    const float* __restrict__ hin, const long long* __restrict__ ei,
    const float* __restrict__ eattr, const float* __restrict__ ew,
    const float* __restrict__ ebv, float* __restrict__ agg, int nE) {
  int e = blockIdx.x * 8 + (threadIdx.x >> 5);
  if (e >= nE) return;                         // wave-uniform exit
  int lane = threadIdx.x & 31;
  long long s = ei[e];
  long long d = ei[(size_t)nE + e];
  float a = eattr[e];
  int c = lane << 2;
  float4 hv = *(const float4*)(hin + (size_t)s * H + c);
  float4 wv = *(const float4*)(ew + c);
  float4 bv = *(const float4*)(ebv + c);
  float m0 = fmaxf(hv.x + a * wv.x + bv.x, 0.0f);
  float m1 = fmaxf(hv.y + a * wv.y + bv.y, 0.0f);
  float m2 = fmaxf(hv.z + a * wv.z + bv.z, 0.0f);
  float m3 = fmaxf(hv.w + a * wv.w + bv.w, 0.0f);
  float* ap = agg + (size_t)d * H + c;
  atomicAdd(ap + 0, m0);
  atomicAdd(ap + 1, m1);
  atomicAdd(ap + 2, m2);
  atomicAdd(ap + 3, m3);
}

// ---------------------------------------------------------------------------
// Fused GIN MLP layer: z=(1+eps)h+agg ; y=relu(z@W1+b1) ; z2=y@W2+b2 ;
// h=relu(gamma*z2*inv_std+beta). 4 waves, 64-row tile, WMMA bf16 f32-acc.
// ---------------------------------------------------------------------------
__global__ __launch_bounds__(128) void gine_layer_kernel(
    const float* __restrict__ hin, const float* __restrict__ agg,
    const float* __restrict__ eps_l,
    const __bf16* __restrict__ w1t, const float* __restrict__ b1,
    const __bf16* __restrict__ w2t, const float* __restrict__ b2,
    const float* __restrict__ gam, const float* __restrict__ bet,
    float* __restrict__ hout, int nN) {
  __shared__ __bf16 sZ[64 * H];   // 16 KB  (z tile, bf16, swizzled)
  __shared__ __bf16 sY[64 * H];   // 16 KB  (gemm1 output, bf16, swizzled)
  __shared__ __bf16 sW[H * H];    // 32 KB  (W^T tile, reused W1 then W2)

  const int tid = threadIdx.x;
  const int rowBase = blockIdx.x * 64;
  const float ep = 1.0f + *eps_l;
  const float inv_std = rsqrtf(1.0f + 1e-5f);

  // Prefetch W2^T toward the caches while GEMM1 runs (global_prefetch_b8).
  for (int i = tid; i < (H * H * 2) / 128; i += 128)
    __builtin_prefetch((const char*)w2t + i * 128, 0, 0);

  // ---- stage W1^T (async DMA to LDS when available) ----
  stage_weights(w1t, sW, tid);

  // ---- stage Z = (1+eps)*h + agg into LDS as bf16 ----
  for (int i = tid; i < 64 * (H / 4); i += 128) {
    int r = i >> 5;                      // 32 float4 per row
    int c = (i & 31) << 2;
    int node = rowBase + r;
    float4 hv = make_float4(0.f, 0.f, 0.f, 0.f);
    float4 av = make_float4(0.f, 0.f, 0.f, 0.f);
    if (node < nN) {
      hv = *(const float4*)(hin + (size_t)node * H + c);
      av = *(const float4*)(agg + (size_t)node * H + c);
    }
    union { uint2 u; __bf16 b[4]; } pk;
    pk.b[0] = (__bf16)(ep * hv.x + av.x);
    pk.b[1] = (__bf16)(ep * hv.y + av.y);
    pk.b[2] = (__bf16)(ep * hv.z + av.z);
    pk.b[3] = (__bf16)(ep * hv.w + av.w);
    *(uint2*)&sZ[r * H + swz(r, c)] = pk.u;
  }
  __syncthreads();

  const int wave = tid >> 5, lane = tid & 31;
  const int half = lane >> 4, lm = lane & 15;
  const int mrow = wave * 16 + lm;     // A-matrix row this lane supplies

  // ================= GEMM1: Y = relu(Z @ W1 + b1) =================
  {
    ABu a[4];
#pragma unroll
    for (int kb = 0; kb < 4; ++kb)
#pragma unroll
      for (int j = 0; j < 8; ++j) {
        // 16-bit A 16x32 layout: VGPR j -> K = 16*(j/4) + 8*half + 2*(j%4)
        int k = kb * 32 + ((j >> 2) << 4) + (half << 3) + ((j & 3) << 1);
        a[kb].u[j] = *(const unsigned int*)&sZ[mrow * H + swz(mrow, k)];
      }
#pragma unroll
    for (int nt = 0; nt < 8; ++nt) {
      v8f acc = {};
      int n = nt * 16 + lm;
#pragma unroll
      for (int kb = 0; kb < 4; ++kb) {
        ABu b;
#pragma unroll
        for (int j = 0; j < 8; ++j) {
          // 16-bit B 32x16 layout: VGPR j -> K = 16*(j/4) + 8*half + 2*(j%4)
          int k = kb * 32 + ((j >> 2) << 4) + (half << 3) + ((j & 3) << 1);
          b.u[j] = *(const unsigned int*)&sW[n * H + swz(n, k)];
        }
        acc = __builtin_amdgcn_wmma_f32_16x16x32_bf16(
            false, a[kb].v, false, b.v, (short)0, acc, false, false);
      }
      float bias = b1[n];
#pragma unroll
      for (int r = 0; r < 8; ++r) {
        // C/D layout: VGPR r holds M = r + 8*half, N = lane%16
        int m = wave * 16 + (half << 3) + r;
        float v = acc[r] + bias;
        v = v > 0.0f ? v : 0.0f;
        sY[m * H + swz(m, n)] = (__bf16)v;
      }
    }
  }
  __syncthreads();
  // ---- swap in W2^T ----
  stage_weights(w2t, sW, tid);
  __syncthreads();

  // ============ GEMM2 + bias + BN(eval) + ReLU, write back ============
  {
    ABu a[4];
#pragma unroll
    for (int kb = 0; kb < 4; ++kb)
#pragma unroll
      for (int j = 0; j < 8; ++j) {
        int k = kb * 32 + ((j >> 2) << 4) + (half << 3) + ((j & 3) << 1);
        a[kb].u[j] = *(const unsigned int*)&sY[mrow * H + swz(mrow, k)];
      }
#pragma unroll
    for (int nt = 0; nt < 8; ++nt) {
      v8f acc = {};
      int n = nt * 16 + lm;
#pragma unroll
      for (int kb = 0; kb < 4; ++kb) {
        ABu b;
#pragma unroll
        for (int j = 0; j < 8; ++j) {
          int k = kb * 32 + ((j >> 2) << 4) + (half << 3) + ((j & 3) << 1);
          b.u[j] = *(const unsigned int*)&sW[n * H + swz(n, k)];
        }
        acc = __builtin_amdgcn_wmma_f32_16x16x32_bf16(
            false, a[kb].v, false, b.v, (short)0, acc, false, false);
      }
      float bias = b2[n];
      float g = gam[n], be = bet[n];
#pragma unroll
      for (int r = 0; r < 8; ++r) {
        int node = rowBase + wave * 16 + (half << 3) + r;
        float v = acc[r] + bias;
        v = g * v * inv_std + be;
        v = v > 0.0f ? v : 0.0f;
        if (node < nN) hout[(size_t)node * H + n] = v;
      }
    }
  }
}

// ---------------------------------------------------------------------------
// Global mean pool: atomic sums + counts, division folded into classifier.
// ---------------------------------------------------------------------------
__global__ __launch_bounds__(256) void pool_sum_kernel(
    const float* __restrict__ hin, const long long* __restrict__ batch,
    float* __restrict__ psum, float* __restrict__ pcnt, int nN) {
  int t = blockIdx.x * 256 + threadIdx.x;
  int node = t >> 5;
  if (node >= nN) return;
  int lane = t & 31;
  int g = (int)batch[node];
  int c = lane << 2;
  float4 hv = *(const float4*)(hin + (size_t)node * H + c);
  float* p = psum + g * H + c;
  atomicAdd(p + 0, hv.x);
  atomicAdd(p + 1, hv.y);
  atomicAdd(p + 2, hv.z);
  atomicAdd(p + 3, hv.w);
  if (lane == 0) atomicAdd(&pcnt[g], 1.0f);
}

__global__ void cls1_kernel(const float* __restrict__ psum,
                            const float* __restrict__ pcnt,
                            const float* __restrict__ cw1,
                            const float* __restrict__ cb1,
                            float* __restrict__ hc, int nG) {
  int t = blockIdx.x * blockDim.x + threadIdx.x;
  if (t >= nG * (H / 2)) return;
  int g = t >> 6, j = t & 63;
  float cnt = fmaxf(pcnt[g], 1.0f);
  float s = 0.0f;
#pragma unroll 8
  for (int k = 0; k < H; ++k) s += psum[g * H + k] * cw1[k * (H / 2) + j];
  hc[t] = fmaxf(s / cnt + cb1[j], 0.0f);
}

__global__ void cls2_kernel(const float* __restrict__ hc,
                            const float* __restrict__ cw2,
                            const float* __restrict__ cb2,
                            float* __restrict__ out, int nG) {
  int g = blockIdx.x * blockDim.x + threadIdx.x;
  if (g >= nG) return;
  float s = 0.0f;
#pragma unroll 8
  for (int j = 0; j < H / 2; ++j) s += hc[g * (H / 2) + j] * cw2[j];
  out[g] = 1.0f / (1.0f + expf(-(s + cb2[0])));
}

// ---------------------------------------------------------------------------
extern "C" void kernel_launch(void* const* d_in, const int* in_sizes, int n_in,
                              void* d_out, int out_size, void* d_ws, size_t ws_size,
                              hipStream_t stream) {
  const float*     x      = (const float*)d_in[0];
  const long long* ei     = (const long long*)d_in[1];   // int64 per reference
  const float*     eattr  = (const float*)d_in[2];
  const long long* batch  = (const long long*)d_in[3];
  const float*     enc_w  = (const float*)d_in[4];
  const float*     enc_b  = (const float*)d_in[5];
  const float*     eenc_w = (const float*)d_in[6];
  const float*     eenc_b = (const float*)d_in[7];
  const float*     eps    = (const float*)d_in[8];
  const float*     w1     = (const float*)d_in[9];
  const float*     b1     = (const float*)d_in[10];
  const float*     w2     = (const float*)d_in[11];
  const float*     b2     = (const float*)d_in[12];
  const float*     gamma  = (const float*)d_in[13];
  const float*     beta   = (const float*)d_in[14];
  const float*     cw1    = (const float*)d_in[15];
  const float*     cb1    = (const float*)d_in[16];
  const float*     cw2    = (const float*)d_in[17];
  const float*     cb2    = (const float*)d_in[18];
  float* out = (float*)d_out;

  const int nN = in_sizes[0];   // N  (x is [N,1])
  const int nE = in_sizes[2];   // E  (edge_attr is [E,1])
  const int nG = out_size;      // G
  const int L  = in_sizes[8];   // layers

  auto al = [](size_t v) { return (v + 255) & ~(size_t)255; };
  char* ws = (char*)d_ws;
  size_t oH  = 0;
  size_t oA  = al(oH + (size_t)nN * H * sizeof(float));
  size_t oW  = al(oA + (size_t)nN * H * sizeof(float));
  size_t oP  = al(oW + (size_t)2 * L * H * H * sizeof(__bf16));
  size_t oC  = oP + (size_t)nG * H * sizeof(float);   // pcnt right after psum
  size_t oHc = al(oC + (size_t)nG * sizeof(float));
  float*  hbuf = (float*)(ws + oH);
  float*  aggb = (float*)(ws + oA);
  __bf16* wt   = (__bf16*)(ws + oW);
  float*  psum = (float*)(ws + oP);
  float*  pcnt = (float*)(ws + oC);
  float*  hc   = (float*)(ws + oHc);

  // node encoder
  {
    long long tot = (long long)nN * H;
    encode_kernel<<<(int)((tot + 255) / 256), 256, 0, stream>>>(x, enc_w, enc_b,
                                                                hbuf, nN);
  }
  // bf16 transposed weights (cheap, deterministic, graph-capture safe)
  prep_w_kernel<<<(2 * L * H * H + 255) / 256, 256, 0, stream>>>(w1, w2, wt, L);

  for (int l = 0; l < L; ++l) {
    (void)hipMemsetAsync(aggb, 0, (size_t)nN * H * sizeof(float), stream);
    edge_msg_kernel<<<(nE + 7) / 8, 256, 0, stream>>>(hbuf, ei, eattr, eenc_w,
                                                      eenc_b, aggb, nE);
    gine_layer_kernel<<<(nN + 63) / 64, 128, 0, stream>>>(
        hbuf, aggb, eps + l,
        wt + (size_t)(2 * l) * H * H,     b1 + l * H,
        wt + (size_t)(2 * l + 1) * H * H, b2 + l * H,
        gamma + l * H, beta + l * H, hbuf, nN);
  }

  (void)hipMemsetAsync(psum, 0, (size_t)nG * (H + 1) * sizeof(float), stream);
  pool_sum_kernel<<<(nN * 32 + 255) / 256, 256, 0, stream>>>(hbuf, batch, psum,
                                                             pcnt, nN);
  cls1_kernel<<<(nG * (H / 2) + 255) / 256, 256, 0, stream>>>(psum, pcnt, cw1,
                                                              cb1, hc, nG);
  cls2_kernel<<<1, ((nG + 63) / 64) * 64, 0, stream>>>(hc, cw2, cb2, out, nG);
}
